// MlpwithSOMCosSimilarityModule_8151847928470
// MI455X (gfx1250) — matrix operand order repaired and checked by reading, not compiled, not run on hardware
//
#include <hip/hip_runtime.h>
#include <hip/hip_bf16.h>
#include <math.h>

typedef __bf16 bf16_t;
typedef __attribute__((ext_vector_type(8)))  __bf16 v8bf;
typedef __attribute__((ext_vector_type(16))) __bf16 v16bf;
typedef __attribute__((ext_vector_type(8)))  float  v8f;
typedef __attribute__((ext_vector_type(4)))  unsigned int v4u;
typedef __attribute__((ext_vector_type(8)))  int v8i;
typedef __attribute__((ext_vector_type(4)))  int v4i;

#define L_DIM 128
#define D_DIM 128
#define H_DIM 256
#define NHID  6
#define BSTAGE_LDS_OFF 131072u

union V16U { v16bf v; v8bf h[2]; };

// A-matrix 16x32 bf16 tile (ISA 7.12.2): lanes0-15 row M=lane, K chunks {k0..k0+7, k0+16..k0+23};
// lanes16-31 row M=lane-16, K chunks {k0+8.., k0+24..}
__device__ __forceinline__ v16bf load_tileA(const bf16_t* base, int r0, int k0,
                                            int stride, int lane) {
  int r = r0 + (lane & 15);
  int k = k0 + ((lane >> 4) << 3);
  const bf16_t* p = base + r * stride + k;
  V16U u;
  u.h[0] = *(const v8bf*)(p);
  u.h[1] = *(const v8bf*)(p + 16);
  return u.v;
}

// B-matrix 32x16 bf16 tile: lanes0-15 col N=lane hold K chunks {k0..k0+7, k0+16..k0+23};
// lanes16-31 col N=lane-16 hold K chunks {k0+8.., k0+24..} (matches A)
__device__ __forceinline__ v16bf load_tileB(const bf16_t* base, int n0, int k0,
                                            int stride, int lane) {
  int n = n0 + (lane & 15);
  int k = k0 + ((lane >> 4) << 3);
  const bf16_t* p = base + n * stride + k;
  V16U u;
  u.h[0] = *(const v8bf*)(p);
  u.h[1] = *(const v8bf*)(p + 16);
  return u.v;
}

// TDM: DMA a 128x128 bf16 tile (row stride 256 elems in global) into LDS at
// BSTAGE_LDS_OFF, packed at stride 128. D# per CDNA5 ISA ch.8 (08_async_tensor).
// clang-23 toolchain: 6-arg builtin (g0 v4u, g1 v8i, g2 v4i, g3 v4i, v8i, cpol).
__device__ __forceinline__ void tdm_load_w_tile(const bf16_t* gaddr) {
  unsigned long long ga = (unsigned long long)(const void*)gaddr;
  v4u g0;
  g0[0] = 1u;                                  // count=1, is_restore=0, gather off
  g0[1] = BSTAGE_LDS_OFF;                      // lds_addr (bytes)
  g0[2] = (unsigned int)(ga & 0xFFFFFFFFu);    // global_addr[31:0]
  g0[3] = (unsigned int)((ga >> 32) & 0x1FFFFFFu) | (2u << 30);  // addr[56:32] | type=2
  v8i g1;
  g1[0] = (int)(1u << 16);    // wg_mask=0 (not in cluster), data_size=1 -> 2 bytes
  g1[1] = (int)(256u << 16);  // tensor_dim0=256 (bits[79:48], low 16 here)
  g1[2] = (int)(256u << 16);  // tensor_dim1=256 (bits[111:80], low 16 here)
  g1[3] = (int)(128u << 16);  // tile_dim0=128 (bits[127:112])
  g1[4] = 128;                // tile_dim1=128, tile_dim2=0
  g1[5] = 256;                // tensor_dim0_stride=256 (low 32 of 48)
  g1[6] = 0;                  // stride hi / tensor_dim1_stride (unused, 2D)
  g1[7] = 0;
  v4i g2 = {0, 0, 0, 0};      // dims 2/3 unused (2D tensor)
  v4i g3 = {0, 0, 0, 0};
  v8i g4 = {0, 0, 0, 0, 0, 0, 0, 0};  // extra descriptor group (unused)
  __builtin_amdgcn_tensor_load_to_lds(g0, g1, g2, g3, g4, 0);
}

// Pre-pass: convert Ws (6x256x256 fp32) to bf16 once; removes per-block cvt work.
__global__ __launch_bounds__(256) void cvt_w_kernel(const float* __restrict__ Ws,
                                                    bf16_t* __restrict__ Wb) {
  int i = (blockIdx.x * 256 + threadIdx.x) * 4;
  float4 f = *(const float4*)(Ws + i);
  bf16_t* p = Wb + i;
  p[0] = (bf16_t)f.x; p[1] = (bf16_t)f.y; p[2] = (bf16_t)f.z; p[3] = (bf16_t)f.w;
}

template <bool USE_TDM>
__global__ __launch_bounds__(256) void som_mlp_kernel(
    const float* __restrict__ context, const float* __restrict__ Ws,
    const bf16_t* __restrict__ Wb, const float* __restrict__ bs,
    const float* __restrict__ W_out, const float* __restrict__ b_out,
    float* __restrict__ out) {
  extern __shared__ __align__(16) char smem[];
  bf16_t* bufX = (bf16_t*)smem;                 // 128 x 256 bf16 (64KB): x / layer output
  bf16_t* bufY = (bf16_t*)(smem + 65536);       // 128 x 256 bf16 (64KB): ping-pong / idx / reduce
  bf16_t* shr  = (bf16_t*)(smem + BSTAGE_LDS_OFF);  // 32KB: ent_n (setup) / W stage (MLP)

  const int t    = threadIdx.x;
  const int lane = t & 31;
  const int wv   = t >> 5;
  const int bk   = blockIdx.x;

  // ---------- Phase 1: row norms + bf16 normalize (single global pass) ----------
  {
    const int l = t >> 1;
    const int h = (t & 1) * 64;
    const float4* cp = (const float4*)(context + ((size_t)(bk * 2 + 0) * L_DIM + l) * D_DIM + h);
    const float4* ep = (const float4*)(context + ((size_t)(bk * 2 + 1) * L_DIM + l) * D_DIM + h);

    float4 v[16];
    float s = 0.f;
#pragma unroll
    for (int j = 0; j < 16; ++j) {
      v[j] = cp[j];
      s += v[j].x * v[j].x + v[j].y * v[j].y + v[j].z * v[j].z + v[j].w * v[j].w;
    }
    s += __shfl_xor(s, 1, 32);
    float inv = rsqrtf(s);
    bf16_t* cd = bufX + l * H_DIM + h;  // ctx_n -> left half of x
#pragma unroll
    for (int j = 0; j < 16; ++j) {
      cd[4 * j + 0] = (bf16_t)(v[j].x * inv);
      cd[4 * j + 1] = (bf16_t)(v[j].y * inv);
      cd[4 * j + 2] = (bf16_t)(v[j].z * inv);
      cd[4 * j + 3] = (bf16_t)(v[j].w * inv);
    }
    s = 0.f;
#pragma unroll
    for (int j = 0; j < 16; ++j) {
      v[j] = ep[j];
      s += v[j].x * v[j].x + v[j].y * v[j].y + v[j].z * v[j].z + v[j].w * v[j].w;
    }
    s += __shfl_xor(s, 1, 32);
    inv = rsqrtf(s);
    bf16_t* ed = shr + l * D_DIM + h;   // ent_n
#pragma unroll
    for (int j = 0; j < 16; ++j) {
      ed[4 * j + 0] = (bf16_t)(v[j].x * inv);
      ed[4 * j + 1] = (bf16_t)(v[j].y * inv);
      ed[4 * j + 2] = (bf16_t)(v[j].z * inv);
      ed[4 * j + 3] = (bf16_t)(v[j].w * inv);
    }
  }
  __syncthreads();

  v8f zero = {};

  // ---------- Phase 2: sim = ctx_n @ ent_n^T via WMMA, argmax fused in registers ----------
  {
    v16bf as[4];
#pragma unroll
    for (int kc = 0; kc < 4; ++kc)
      as[kc] = load_tileA(bufX, wv * 16, kc * 32, H_DIM, lane);

    float bestV[8];
    int   bestI[8];
#pragma unroll
    for (int r = 0; r < 8; ++r) { bestV[r] = -3.4e38f; bestI[r] = 0; }

#pragma unroll
    for (int nb = 0; nb < 8; ++nb) {
      v8f c = zero;
#pragma unroll
      for (int kc = 0; kc < 4; ++kc) {
        v16bf bb = load_tileB(shr, nb * 16, kc * 32, D_DIM, lane);
        c = __builtin_amdgcn_wmma_f32_16x16x32_bf16(false, as[kc], false, bb,
                                                    (short)0, c, false, false);
      }
      int col = nb * 16 + (lane & 15);
#pragma unroll
      for (int r = 0; r < 8; ++r)
        if (c[r] > bestV[r]) { bestV[r] = c[r]; bestI[r] = col; }
    }
#pragma unroll
    for (int off = 1; off < 16; off <<= 1) {
#pragma unroll
      for (int r = 0; r < 8; ++r) {
        float ov = __shfl_xor(bestV[r], off, 32);
        int   oi = __shfl_xor(bestI[r], off, 32);
        if (ov > bestV[r]) { bestV[r] = ov; bestI[r] = oi; }
      }
    }
    int* idxArr = (int*)bufY;  // bufY dead until MLP layer 0 writes it
    if ((lane & 15) == 0) {
      int rb = wv * 16 + ((lane >> 4) << 3);
#pragma unroll
      for (int r = 0; r < 8; ++r) idxArr[rb + r] = bestI[r];
    }
  }
  __syncthreads();

  // ---------- Phase 3: gather ent_n[argmax] into right half of x ----------
  {
    int l = t >> 1, h = (t & 1) * 64;
    int sr = ((const int*)bufY)[l];
    const v8bf* sp = (const v8bf*)(shr + sr * D_DIM + h);
    v8bf* dp = (v8bf*)(bufX + l * H_DIM + D_DIM + h);
#pragma unroll
    for (int j = 0; j < 8; ++j) dp[j] = sp[j];
  }

  // ---------- Phase 4: 6-layer MLP, y = tanh(x @ W^T + b) ----------
  bf16_t* Bstage = shr;  // 128 x 128 bf16 quarter of W, n-major (stride 128)
  for (int layer = 0; layer < NHID; ++layer) {
    const bf16_t* src = (layer & 1) ? bufY : bufX;
    bf16_t*       dst = (layer & 1) ? bufX : bufY;
    const float* bl = bs + layer * H_DIM;
    for (int np = 0; np < 2; ++np) {         // output-column halves
      v8f acc[8];
#pragma unroll
      for (int n = 0; n < 8; ++n) acc[n] = zero;
      for (int kb = 0; kb < 2; ++kb) {       // K halves
        __syncthreads();  // previous stage consumers done
        if (USE_TDM) {
          // TDM DMA: one descriptor moves the whole 32KB quarter; wave 0 waits for it before the barrier.
          if (wv == 0) {
            const bf16_t* gsrc = Wb + (size_t)layer * H_DIM * H_DIM +
                                 (size_t)(np * 128) * H_DIM + kb * 128;
            tdm_load_w_tile(gsrc);
            __builtin_amdgcn_s_wait_tensorcnt(0);
          }
        } else {
          const float* Wl = Ws + layer * H_DIM * H_DIM;
          int nl = t >> 1;
          int kl = (t & 1) * 64;
          const float4* wp = (const float4*)(Wl + (size_t)(np * 128 + nl) * H_DIM + kb * 128 + kl);
          bf16_t* bp = Bstage + nl * 128 + kl;
#pragma unroll
          for (int j = 0; j < 16; ++j) {
            float4 f = wp[j];
            bp[4 * j + 0] = (bf16_t)f.x;
            bp[4 * j + 1] = (bf16_t)f.y;
            bp[4 * j + 2] = (bf16_t)f.z;
            bp[4 * j + 3] = (bf16_t)f.w;
          }
        }
        __syncthreads();  // staging visible to all waves
#pragma unroll
        for (int kc = 0; kc < 4; ++kc) {
          v16bf a = load_tileA(src, wv * 16, kb * 128 + kc * 32, H_DIM, lane);
#pragma unroll
          for (int n = 0; n < 8; ++n) {
            v16bf bb = load_tileB(Bstage, n * 16, kc * 32, 128, lane);
            acc[n] = __builtin_amdgcn_wmma_f32_16x16x32_bf16(false, a, false, bb,
                                                             (short)0, acc[n], false, false);
          }
        }
      }
      // epilogue: bias + tanh + bf16 store (C/D layout: VGPR r -> row r / r+8 per half)
      int rb = wv * 16 + ((lane >> 4) << 3);
#pragma unroll
      for (int n = 0; n < 8; ++n) {
        int col = np * 128 + n * 16 + (lane & 15);
        float bv = bl[col];
#pragma unroll
        for (int r = 0; r < 8; ++r)
          dst[(rb + r) * H_DIM + col] = (bf16_t)tanhf(acc[n][r] + bv);
      }
    }
  }
  __syncthreads();

  // ---------- Phase 5: sum_l (x @ W_out) + L*b_out (final x in bufX) ----------
  {
    float s = 0.f;
#pragma unroll 8
    for (int r = 0; r < L_DIM; ++r) s += (float)bufX[r * H_DIM + t];
    s *= W_out[t];
#pragma unroll
    for (int off = 1; off < 32; off <<= 1) s += __shfl_xor(s, off, 32);
    float* red = (float*)bufY;
    if (lane == 0) red[wv] = s;
    __syncthreads();
    if (t == 0) {
      float tot = 0.f;
#pragma unroll
      for (int w = 0; w < 8; ++w) tot += red[w];
      out[bk] = tot + (float)L_DIM * b_out[0];
    }
  }
}

extern "C" void kernel_launch(void* const* d_in, const int* in_sizes, int n_in,
                              void* d_out, int out_size, void* d_ws, size_t ws_size,
                              hipStream_t stream) {
  const float* context = (const float*)d_in[0];
  const float* Ws      = (const float*)d_in[1];
  const float* bs      = (const float*)d_in[2];
  const float* W_out   = (const float*)d_in[3];
  const float* b_out   = (const float*)d_in[4];
  float* out = (float*)d_out;

  const size_t lds = 160 * 1024;  // 64K x + 64K y + 32K ent/W-stage
  const size_t wbytes = (size_t)NHID * H_DIM * H_DIM * sizeof(bf16_t);

  if (ws_size >= wbytes) {
    bf16_t* Wb = (bf16_t*)d_ws;
    // 6*256*256 / (256*4) = 384 blocks
    cvt_w_kernel<<<dim3((NHID * H_DIM * H_DIM) / 1024), dim3(256), 0, stream>>>(Ws, Wb);
    (void)hipFuncSetAttribute((const void*)som_mlp_kernel<true>,
                              hipFuncAttributeMaxDynamicSharedMemorySize, (int)lds);
    som_mlp_kernel<true><<<dim3(64 * 64), dim3(256), lds, stream>>>(
        context, Ws, Wb, bs, W_out, b_out, out);
  } else {
    (void)hipFuncSetAttribute((const void*)som_mlp_kernel<false>,
                              hipFuncAttributeMaxDynamicSharedMemorySize, (int)lds);
    som_mlp_kernel<false><<<dim3(64 * 64), dim3(256), lds, stream>>>(
        context, Ws, nullptr, bs, W_out, b_out, out);
  }
}
